// ImprovedGAT_44238163149266
// MI455X (gfx1250) — matrix operand order repaired and checked by reading, not compiled, not run on hardware
//
#include <hip/hip_runtime.h>
#include <math.h>

#define IN_CH  64
#define HID    96
#define HEADS  4
#define CPH    24          // channels per head
#define LAYERS 4
#define BN_EPS 1e-5f
#define MT     4           // 16-row tiles per block (64 rows/block)

typedef __attribute__((ext_vector_type(16))) _Float16 v16h;
typedef __attribute__((ext_vector_type(8)))  float    v8f;

#if __has_builtin(__builtin_amdgcn_global_load_async_to_lds_b128)
#define USE_ASYNC_LDS 1
#else
#define USE_ASYNC_LDS 0
#endif

#if USE_ASYNC_LDS
// async-to-LDS builtin takes (int4 AS1*, int4 AS3*, imm offset, imm cpol)
typedef int async_i4 __attribute__((vector_size(16)));
typedef __attribute__((address_space(1))) async_i4* g_i4p;
typedef __attribute__((address_space(3))) async_i4* l_i4p;
typedef __attribute__((address_space(1))) void*     g_vp;
typedef __attribute__((address_space(3))) void*     l_vp;
#endif

__device__ __forceinline__ void wait_async_all() {
#if USE_ASYNC_LDS
#if __has_builtin(__builtin_amdgcn_s_wait_asynccnt)
    __builtin_amdgcn_s_wait_asynccnt(0);
#else
    asm volatile("s_wait_asynccnt 0x0" ::: "memory");
#endif
#endif
}

__device__ __forceinline__ float elu1(float x)  { return x > 0.f ? x : __expf(x) - 1.f; }
__device__ __forceinline__ float lrelu(float x) { return x > 0.f ? x : 0.2f * x; }

// float atomic max via sign-split integer atomics (monotone mapping)
__device__ __forceinline__ void atomicMaxF(float* addr, float v) {
    if (v >= 0.f) atomicMax((int*)addr, __float_as_int(v));
    else          atomicMin((unsigned int*)addr, __float_as_uint(v));
}

// ---------------------------------------------------------------------------
// WMMA GEMM: C[M x 96] = act(A[M x K] @ W[K x 96] + bias)
// block = 192 threads = 6 wave32; wave w owns output columns [16w, 16w+16)
// block covers MT*16 = 64 rows; K compile-time (multiple of 32).
// B fragments preloaded to registers; A double-buffered through LDS
// via gfx1250 GLOBAL_LOAD_ASYNC_TO_LDS_B128 (fallback: float4 via VGPRs).
// ACT: 0 = none, 1 = ELU, 2 = ReLU
// ---------------------------------------------------------------------------
template<int K, int ACT>
__global__ __launch_bounds__(192)
void k_gemm_wmma(const float* __restrict__ A, const float* __restrict__ W,
                 const float* __restrict__ bias, float* __restrict__ C, int M)
{
    constexpr int NC = K / 32;                     // K chunks
    __shared__ __align__(16) float As[2][MT * 16][36];  // pitch 36 -> conflict-free, 16B aligned

    const int tid  = threadIdx.x;
    const int lane = tid & 31;
    const int wave = tid >> 5;                     // 0..5 -> N tile
    const int m0   = blockIdx.x * (MT * 16);

    // ---- preload B fragments (f32 -> f16) straight from global ----
    // B 32x16 operand, contiguous k order used here (same as A): lanes 0-15 col=lane K=0..15; lanes 16-31 K=16..31
    const int bcol = wave * 16 + (lane & 15);
    const int bk0  = (lane < 16) ? 0 : 16;
    v16h bf[NC];
    #pragma unroll
    for (int c = 0; c < NC; ++c) {
        #pragma unroll
        for (int i = 0; i < 16; ++i)
            bf[c][i] = (_Float16)W[(size_t)(c * 32 + bk0 + i) * HID + bcol];
    }

    // ---- A tile staging (64 rows x 32 cols f32, as 8x b128 per row) ----
    auto stage = [&](int kc, int b) {
        for (int i = tid; i < MT * 16 * 8; i += 192) {
            int r  = i >> 3;
            int cp = (i & 7) * 4;
            int gr = m0 + r; if (gr >= M) gr = M - 1;   // clamp OOB rows
            const float* gp = A + (size_t)gr * K + kc + cp;
#if USE_ASYNC_LDS
            __builtin_amdgcn_global_load_async_to_lds_b128(
                (g_i4p)(g_vp)const_cast<float*>(gp),
                (l_i4p)(l_vp)&As[b][r][cp],
                0, 0);
#else
            const float4 v = *(const float4*)gp;
            *(float4*)&As[b][r][cp] = v;
#endif
        }
    };

    stage(0, 0);

    // A 16x32 operand, contiguous k order used here (same as B): lanes 0-15 M=lane, halfs = K0..15;
    // lanes 16-31 M=lane-16, halfs = K16..31
    const int arow = lane & 15;
    const int ak0  = (lane < 16) ? 0 : 16;

    v8f acc[MT];
    #pragma unroll
    for (int mt = 0; mt < MT; ++mt) acc[mt] = (v8f){};

    #pragma unroll
    for (int c = 0; c < NC; ++c) {
        wait_async_all();
        __syncthreads();                           // chunk c visible to all waves
        if (c + 1 < NC) stage((c + 1) * 32, (c + 1) & 1);

        #pragma unroll
        for (int mt = 0; mt < MT; ++mt) {
            const float* bp = &As[c & 1][mt * 16 + arow][0];
            float4 f0 = *(const float4*)(bp + ak0);
            float4 f1 = *(const float4*)(bp + ak0 + 4);
            float4 f2 = *(const float4*)(bp + ak0 + 8);
            float4 f3 = *(const float4*)(bp + ak0 + 12);
            v16h av;
            av[0]  = (_Float16)f0.x; av[1]  = (_Float16)f0.y;
            av[2]  = (_Float16)f0.z; av[3]  = (_Float16)f0.w;
            av[4]  = (_Float16)f1.x; av[5]  = (_Float16)f1.y;
            av[6]  = (_Float16)f1.z; av[7]  = (_Float16)f1.w;
            av[8]  = (_Float16)f2.x; av[9]  = (_Float16)f2.y;
            av[10] = (_Float16)f2.z; av[11] = (_Float16)f2.w;
            av[12] = (_Float16)f3.x; av[13] = (_Float16)f3.y;
            av[14] = (_Float16)f3.z; av[15] = (_Float16)f3.w;
            acc[mt] = __builtin_amdgcn_wmma_f32_16x16x32_f16(
                          false, av, false, bf[c], (short)0, acc[mt], false, false);
        }
    }

    // C/D layout: VGPR r -> row r (lanes 0-15) / row r+8 (lanes 16-31), col = lane%16
    const int col  = wave * 16 + (lane & 15);
    const float bb = bias ? bias[col] : 0.f;
    #pragma unroll
    for (int mt = 0; mt < MT; ++mt) {
        if (m0 + mt * 16 < M) {                    // whole 16-tile valid (M % 16 == 0)
            const int rb = m0 + mt * 16 + ((lane < 16) ? 0 : 8);
            #pragma unroll
            for (int r = 0; r < 8; ++r) {
                float v = acc[mt][r] + bb;
                if (ACT == 1)      v = elu1(v);
                else if (ACT == 2) v = fmaxf(v, 0.f);
                C[(size_t)(rb + r) * HID + col] = v;
            }
        }
    }
}

// ---------------------------------------------------------------------------
// per-(node, head) attention logits  s = <h[n,h,:], att[h,:]>
// ---------------------------------------------------------------------------
__global__ void k_scores(const float* __restrict__ hn,
                         const float* __restrict__ asrc, const float* __restrict__ adst,
                         float* __restrict__ ssrc, float* __restrict__ sdst, int Nn)
{
    int idx = blockIdx.x * blockDim.x + threadIdx.x;
    if (idx >= Nn * HEADS) return;
    int n = idx >> 2, h = idx & 3;
    const float* row = hn + (size_t)n * HID + h * CPH;
    float s = 0.f, d = 0.f;
    #pragma unroll
    for (int c = 0; c < CPH; ++c) {
        float v = row[c];
        s += v * asrc[h * CPH + c];
        d += v * adst[h * CPH + c];
    }
    ssrc[idx] = s;
    sdst[idx] = d;
}

__global__ void k_fill(float* __restrict__ p, float v, int n)
{
    int i = blockIdx.x * blockDim.x + threadIdx.x;
    if (i < n) p[i] = v;
}

// edge helpers: logical edge list = E real edges + N self loops
__device__ __forceinline__ void edge_sd(const int* ei, int E, int e, int& s, int& d)
{
    if (e < E) { s = ei[e]; d = ei[E + e]; }
    else       { s = e - E; d = e - E; }
}

__global__ void k_edge_max(const int* __restrict__ ei, int E, int Nn,
                           const float* __restrict__ ssrc, const float* __restrict__ sdst,
                           float* __restrict__ mx)
{
    int idx = blockIdx.x * blockDim.x + threadIdx.x;
    int tot = (E + Nn) * HEADS;
    if (idx >= tot) return;
    int e = idx >> 2, h = idx & 3, s, d;
    edge_sd(ei, E, e, s, d);
    float v = lrelu(ssrc[s * HEADS + h] + sdst[d * HEADS + h]);
    atomicMaxF(&mx[d * HEADS + h], v);
}

__global__ void k_edge_sum(const int* __restrict__ ei, int E, int Nn,
                           const float* __restrict__ ssrc, const float* __restrict__ sdst,
                           const float* __restrict__ mx, float* __restrict__ zs)
{
    int idx = blockIdx.x * blockDim.x + threadIdx.x;
    int tot = (E + Nn) * HEADS;
    if (idx >= tot) return;
    int e = idx >> 2, h = idx & 3, s, d;
    edge_sd(ei, E, e, s, d);
    float v = lrelu(ssrc[s * HEADS + h] + sdst[d * HEADS + h]);
    atomicAdd(&zs[d * HEADS + h], __expf(v - mx[d * HEADS + h]));
}

__global__ void k_edge_agg(const int* __restrict__ ei, int E, int Nn,
                           const float* __restrict__ ssrc, const float* __restrict__ sdst,
                           const float* __restrict__ mx, const float* __restrict__ zs,
                           const float* __restrict__ hn, float* __restrict__ agg)
{
    int idx = blockIdx.x * blockDim.x + threadIdx.x;
    int tot = (E + Nn) * HEADS;
    if (idx >= tot) return;
    int e = idx >> 2, h = idx & 3, s, d;
    edge_sd(ei, E, e, s, d);
    float v = lrelu(ssrc[s * HEADS + h] + sdst[d * HEADS + h]);
    float alpha = __expf(v - mx[d * HEADS + h]) / (zs[d * HEADS + h] + 1e-16f);
    const float* srow = hn  + (size_t)s * HID + h * CPH;
    float*       drow = agg + (size_t)d * HID + h * CPH;
    #pragma unroll
    for (int c = 0; c < CPH; ++c)
        atomicAdd(&drow[c], alpha * srow[c]);
}

// bias + BatchNorm(eval) + ELU + residual into h
__global__ void k_bn_res(const float* __restrict__ agg, const float* __restrict__ bg,
                         const float* __restrict__ gamma, const float* __restrict__ beta,
                         const float* __restrict__ rm, const float* __restrict__ rv,
                         float* __restrict__ h, int Nn)
{
    int idx = blockIdx.x * blockDim.x + threadIdx.x;
    if (idx >= Nn * HID) return;
    int f = idx % HID;
    float v = agg[idx] + bg[f];
    v = (v - rm[f]) * rsqrtf(rv[f] + BN_EPS) * gamma[f] + beta[f];
    h[idx] += elu1(v);
}

// graph pooling: pooled[batch[n], layer*HID + f] += h[n, f]
__global__ void k_pool(const float* __restrict__ h, const int* __restrict__ batch,
                       float* __restrict__ pooled, int layer, int Nn)
{
    int idx = blockIdx.x * blockDim.x + threadIdx.x;
    if (idx >= Nn * HID) return;
    int n = idx / HID, f = idx % HID;
    atomicAdd(&pooled[(size_t)batch[n] * (HID * LAYERS) + layer * HID + f], h[idx]);
}

// final head: out[g] = z2[g,:] @ W_h2 + b_h2
__global__ void k_head(const float* __restrict__ z2, const float* __restrict__ W2,
                       const float* __restrict__ b2, float* __restrict__ out, int Gn)
{
    int g = blockIdx.x * blockDim.x + threadIdx.x;
    if (g >= Gn) return;
    float acc = b2[0];
    #pragma unroll
    for (int f = 0; f < HID; ++f) acc += z2[(size_t)g * HID + f] * W2[f];
    out[g] = acc;
}

// ---------------------------------------------------------------------------
extern "C" void kernel_launch(void* const* d_in, const int* in_sizes, int n_in,
                              void* d_out, int out_size, void* d_ws, size_t ws_size,
                              hipStream_t stream)
{
    const float* x       = (const float*)d_in[0];
    const int*   ei      = (const int*)  d_in[1];
    const int*   batch   = (const int*)  d_in[2];
    const float* W_in    = (const float*)d_in[3];
    const float* b_in    = (const float*)d_in[4];
    const float* Wg      = (const float*)d_in[5];   // [L,96,96]
    const float* att_src = (const float*)d_in[6];   // [L,4,24]
    const float* att_dst = (const float*)d_in[7];
    const float* bg      = (const float*)d_in[8];   // [L,96]
    const float* gamma   = (const float*)d_in[9];
    const float* beta    = (const float*)d_in[10];
    const float* rm      = (const float*)d_in[11];
    const float* rv      = (const float*)d_in[12];
    const float* W_jk    = (const float*)d_in[13];  // [384,96]
    const float* b_jk    = (const float*)d_in[14];
    const float* W_h1    = (const float*)d_in[15];  // [96,96]
    const float* b_h1    = (const float*)d_in[16];
    const float* W_h2    = (const float*)d_in[17];  // [96,1]
    const float* b_h2    = (const float*)d_in[18];
    float*       out     = (float*)d_out;

    const int N = in_sizes[0] / IN_CH;   // 50000 (multiple of 16)
    const int E = in_sizes[1] / 2;       // 800000
    const int G = out_size;              // 1024

    // workspace carve-out (floats)
    float* ws     = (float*)d_ws;
    float* h      = ws;  ws += (size_t)N * HID;
    float* hn     = ws;  ws += (size_t)N * HID;
    float* agg    = ws;  ws += (size_t)N * HID;
    float* ssrc   = ws;  ws += (size_t)N * HEADS;
    float* sdst   = ws;  ws += (size_t)N * HEADS;
    float* mx     = ws;  ws += (size_t)N * HEADS;
    float* zs     = ws;  ws += (size_t)N * HEADS;
    float* pooled = ws;  ws += (size_t)G * HID * LAYERS;
    float* z1     = ws;  ws += (size_t)G * HID;
    float* z2     = ws;  ws += (size_t)G * HID;

    const int nh   = N * HID;
    const int nhd  = N * HEADS;
    const int etot = (E + N) * HEADS;
    const int TB   = 256;
    dim3 blk(TB);
    auto cdiv = [](int a, int b) { return (a + b - 1) / b; };

    // h = elu(x @ W_in + b_in)
    k_gemm_wmma<IN_CH, 1><<<dim3(cdiv(N, MT * 16)), dim3(192), 0, stream>>>(x, W_in, b_in, h, N);

    // zero pooled once
    k_fill<<<dim3(cdiv(G * HID * LAYERS, TB)), blk, 0, stream>>>(pooled, 0.f, G * HID * LAYERS);

    for (int l = 0; l < LAYERS; ++l) {
        const float* Wl  = Wg      + (size_t)l * HID * HID;
        const float* asl = att_src + (size_t)l * HEADS * CPH;
        const float* adl = att_dst + (size_t)l * HEADS * CPH;

        // hn = h @ Wg[l]   (WMMA)
        k_gemm_wmma<HID, 0><<<dim3(cdiv(N, MT * 16)), dim3(192), 0, stream>>>(h, Wl, nullptr, hn, N);

        // attention logits per node/head
        k_scores<<<dim3(cdiv(nhd, TB)), blk, 0, stream>>>(hn, asl, adl, ssrc, sdst, N);

        // init segment max / sum / aggregate buffers
        k_fill<<<dim3(cdiv(nhd, TB)), blk, 0, stream>>>(mx, -3.0e38f, nhd);
        k_fill<<<dim3(cdiv(nhd, TB)), blk, 0, stream>>>(zs, 0.f, nhd);
        k_fill<<<dim3(cdiv(nh,  TB)), blk, 0, stream>>>(agg, 0.f, nh);

        // segment softmax (3 passes over E + N self-loop edges)
        k_edge_max<<<dim3(cdiv(etot, TB)), blk, 0, stream>>>(ei, E, N, ssrc, sdst, mx);
        k_edge_sum<<<dim3(cdiv(etot, TB)), blk, 0, stream>>>(ei, E, N, ssrc, sdst, mx, zs);
        k_edge_agg<<<dim3(cdiv(etot, TB)), blk, 0, stream>>>(ei, E, N, ssrc, sdst, mx, zs, hn, agg);

        // bias + BN + ELU + residual
        k_bn_res<<<dim3(cdiv(nh, TB)), blk, 0, stream>>>(
            agg, bg + (size_t)l * HID, gamma + (size_t)l * HID, beta + (size_t)l * HID,
            rm + (size_t)l * HID, rv + (size_t)l * HID, h, N);

        // pooled[:, l*96:(l+1)*96] += segment_sum(h, batch)
        k_pool<<<dim3(cdiv(nh, TB)), blk, 0, stream>>>(h, batch, pooled, l, N);
    }

    // z1 = elu(pooled @ W_jk + b_jk)   [G x 384] @ [384 x 96]
    k_gemm_wmma<HID * LAYERS, 1><<<dim3(cdiv(G, MT * 16)), dim3(192), 0, stream>>>(pooled, W_jk, b_jk, z1, G);
    // z2 = relu(z1 @ W_h1 + b_h1)
    k_gemm_wmma<HID, 2><<<dim3(cdiv(G, MT * 16)), dim3(192), 0, stream>>>(z1, W_h1, b_h1, z2, G);
    // out = z2 @ W_h2 + b_h2
    k_head<<<dim3(cdiv(G, TB)), blk, 0, stream>>>(z2, W_h2, b_h2, out, G);
}
